// RBF_5901285064824
// MI455X (gfx1250) — matrix operand order repaired and checked by reading, not compiled, not run on hardware
//
#include <hip/hip_runtime.h>
#include <hip/hip_bf16.h>

// ---- dtypes for WMMA -------------------------------------------------------
typedef __attribute__((ext_vector_type(16))) __bf16          v16bf;
typedef __attribute__((ext_vector_type(16))) unsigned short  v16us;
typedef __attribute__((ext_vector_type(8)))  unsigned short  v8us;
typedef __attribute__((ext_vector_type(8)))  unsigned int    v8ui;
typedef __attribute__((ext_vector_type(8)))  float           v8f;
typedef __attribute__((ext_vector_type(4)))  float           v4f;

// ---- problem constants -----------------------------------------------------
#define B_ 16384
#define D_ 512
#define C_ 4096
#define H_ 1024

#define BT  32            // x rows per block
#define CB  64            // C-chunk per iteration
#define PLD (CB + 8)      // 72 : phi row stride -> 144 B, 16B-aligned rows

#define LOG2E 1.44269504088896340736f

// ---- workspace layout (bytes) ----------------------------------------------
#define WS_XBF   ((size_t)0)                       // B*D bf16 = 16 MiB
#define WS_CBF   (WS_XBF + (size_t)B_ * D_ * 2)    // C*D bf16 =  4 MiB
#define WS_W1BF  (WS_CBF + (size_t)C_ * D_ * 2)    // H*C bf16 =  8 MiB
#define WS_X2    (WS_W1BF + (size_t)H_ * C_ * 2)   // B f32
#define WS_C2    (WS_X2 + (size_t)B_ * 4)          // C f32
#define WS_NEED  (WS_C2 + (size_t)C_ * 4)          // ~28.1 MiB

// round-to-nearest f32 -> bf16 (ties away), single element
__device__ __forceinline__ unsigned short f2bf(float f) {
    return (unsigned short)((__float_as_uint(f) + 0x8000u) >> 16);
}

// pack two f32 -> packed bf16 pair: 2x v_add + 1x v_perm_b32
__device__ __forceinline__ unsigned pack2bf(float lo, float hi) {
    unsigned ul = __float_as_uint(lo) + 0x8000u;
    unsigned uh = __float_as_uint(hi) + 0x8000u;
    return __builtin_amdgcn_perm(uh, ul, 0x07060302u);
}

__device__ __forceinline__ v16us pack16(v8us lo, v8us hi) {
    v16us r;
#pragma unroll
    for (int i = 0; i < 8; ++i) { r[i] = lo[i]; r[i + 8] = hi[i]; }
    return r;
}

__device__ __forceinline__ v8f wmma_bf16(v16us a, v16us b, v8f c) {
    return __builtin_amdgcn_wmma_f32_16x16x32_bf16(
        false, __builtin_bit_cast(v16bf, a),
        false, __builtin_bit_cast(v16bf, b),
        (short)0, c, false, false);
}

// ============================================================================
// Pre-pass A: row-major [rows, 512] f32 -> bf16 + exact f32 row sum-of-squares
// One wave per row; 8 waves per block.
// ============================================================================
__global__ __launch_bounds__(256, 1)
void conv_rows_kernel(const float* __restrict__ src,
                      unsigned short* __restrict__ dst,
                      float* __restrict__ sumsq) {
    const int wave = threadIdx.x >> 5;
    const int lane = threadIdx.x & 31;
    const int row  = blockIdx.x * 8 + wave;
    const v4f*  s4 = (const v4f*)(src + (size_t)row * D_);
    uint2*      d2 = (uint2*)(dst + (size_t)row * D_);
    float s = 0.0f;
#pragma unroll
    for (int j = 0; j < 4; ++j) {
        v4f v = s4[j * 32 + lane];
        s = fmaf(v[0], v[0], s); s = fmaf(v[1], v[1], s);
        s = fmaf(v[2], v[2], s); s = fmaf(v[3], v[3], s);
        uint2 p; p.x = pack2bf(v[0], v[1]); p.y = pack2bf(v[2], v[3]);
        d2[j * 32 + lane] = p;
    }
#pragma unroll
    for (int off = 16; off > 0; off >>= 1) s += __shfl_xor(s, off, 32);
    if (lane == 0) sumsq[row] = s;
}

// ============================================================================
// Pre-pass B: flat f32 -> bf16 (for W1); 8 elements per thread
// ============================================================================
__global__ __launch_bounds__(256, 1)
void conv_flat_kernel(const float* __restrict__ src,
                      unsigned short* __restrict__ dst) {
    const size_t i = ((size_t)blockIdx.x * 256 + threadIdx.x) * 8;
    v4f a = *(const v4f*)(src + i);
    v4f b = *(const v4f*)(src + i + 4);
    uint4 p;
    p.x = pack2bf(a[0], a[1]); p.y = pack2bf(a[2], a[3]);
    p.z = pack2bf(b[0], b[1]); p.w = pack2bf(b[2], b[3]);
    *(uint4*)(dst + i) = p;
}

// ============================================================================
// Main fused kernel (fast path): operands already bf16 in ws.
// out = sigmoid( exp(-dist(x, centers)/(2*sigma)) @ W1^T + b1 )
// ============================================================================
__global__ __launch_bounds__(256, 1)
void rbf_fused_fast(const unsigned short* __restrict__ xbf,
                    const unsigned short* __restrict__ cbf,
                    const unsigned short* __restrict__ w1bf,
                    const float* __restrict__ x2g,
                    const float* __restrict__ c2g,
                    const float* __restrict__ b1,
                    const int*   __restrict__ sigma_p,
                    float* __restrict__ out) {
    __shared__ unsigned short phis[BT][PLD];   // bf16 phi tile (4.6 KB)

    const int tid   = threadIdx.x;
    const int wave  = tid >> 5;
    const int lane  = tid & 31;
    const int laneN = lane & 15;
    const int hi    = lane >> 4;
    const int m0    = blockIdx.x * BT;

    const float sig      = (float)sigma_p[0];
    const float negScale = -LOG2E * 0.5f / sig;

    const int mt = wave & 1;                // GEMM1 tile owned by this wave
    const int nt = wave >> 1;

    // x^2 values for this wave's 8 accumulator rows: live in registers forever
    float xv[8];
    {
        const float* x2row = x2g + m0 + mt * 16 + hi * 8;
#pragma unroll
        for (int j = 0; j < 8; ++j) xv[j] = x2row[j];
    }

    // GEMM1 fragment base pointers (global bf16, immediate k offsets)
    const unsigned short* arow = xbf + (size_t)(m0 + mt * 16 + laneN) * D_ + hi * 16;
    const unsigned short* brow0 = cbf + (size_t)(nt * 16 + laneN) * D_ + hi * 16;

    // GEMM2 W1 row base pointers (8 n-tiles owned by this wave)
    const unsigned short* w1p[8];
#pragma unroll
    for (int nt2 = 0; nt2 < 8; ++nt2)
        w1p[nt2] = w1bf + (size_t)(wave * 128 + nt2 * 16 + laneN) * C_ + hi * 16;

    // persistent GEMM2 accumulators: 2 m-tiles x 8 n-tiles
    v8f acc2[2][8];
#pragma unroll
    for (int i = 0; i < 2; ++i)
#pragma unroll
        for (int j = 0; j < 8; ++j)
#pragma unroll
            for (int e = 0; e < 8; ++e) acc2[i][j][e] = 0.0f;

    for (int cc = 0; cc < C_; cc += CB) {
        // ---- GEMM1: S[32,64] = x_tile @ centers_chunk^T, K = 512 ----------
        v8f sacc;
#pragma unroll
        for (int e = 0; e < 8; ++e) sacc[e] = 0.0f;
        {
            const unsigned short* brow = brow0 + (size_t)cc * D_;
#pragma unroll 8
            for (int kk = 0; kk < D_; kk += 32) {
                v16us av = pack16(*(const v8us*)(arow + kk),
                                  *(const v8us*)(arow + kk + 8));
                v16us bv = pack16(*(const v8us*)(brow + kk),
                                  *(const v8us*)(brow + kk + 8));
                sacc = wmma_bf16(av, bv, sacc);
            }
        }

        __syncthreads();   // prior iter's GEMM2 done reading phis

        // ---- elementwise: phi = exp2(negScale * sqrt(max(x2+c2-2*xc,0))) --
        {
            const float c2v = c2g[cc + nt * 16 + laneN];
#pragma unroll
            for (int j = 0; j < 8; ++j) {
                const int m = mt * 16 + j + 8 * hi;
                float d2 = fmaxf(fmaf(-2.0f, sacc[j], xv[j] + c2v), 0.0f);
                float d  = __builtin_amdgcn_sqrtf(d2);
                float p  = __builtin_amdgcn_exp2f(d * negScale);
                phis[m][nt * 16 + laneN] = f2bf(p);
            }
        }
        __syncthreads();

        // ---- GEMM2: acc2 += phi[32,64] @ W1_chunk^T (wave owns 128 cols) --
#pragma unroll
        for (int ks = 0; ks < 2; ++ks) {
            v16us afr[2];
#pragma unroll
            for (int mt2 = 0; mt2 < 2; ++mt2) {
                const unsigned short* ar = &phis[mt2 * 16 + laneN][ks * 32 + hi * 16];
                afr[mt2] = pack16(*(const v8us*)ar, *(const v8us*)(ar + 8));
            }
#pragma unroll
            for (int nt2 = 0; nt2 < 8; ++nt2) {
                const v8us* wp = (const v8us*)(w1p[nt2] + cc + ks * 32);
                v16us bfr = pack16(wp[0], wp[1]);
                acc2[0][nt2] = wmma_bf16(afr[0], bfr, acc2[0][nt2]);
                acc2[1][nt2] = wmma_bf16(afr[1], bfr, acc2[1][nt2]);
            }
        }
    }

    // ---- epilogue: sigmoid(acc + b1) -> out --------------------------------
#pragma unroll
    for (int nt2 = 0; nt2 < 8; ++nt2) {
        const int hcol = wave * 128 + nt2 * 16 + laneN;
        const float bb = b1[hcol];
#pragma unroll
        for (int mt2 = 0; mt2 < 2; ++mt2) {
#pragma unroll
            for (int j = 0; j < 8; ++j) {
                const int row = m0 + mt2 * 16 + j + 8 * hi;
                float z = acc2[mt2][nt2][j] + bb;
                float e = __builtin_amdgcn_exp2f(-z * LOG2E);
                out[(size_t)row * H_ + hcol] = __builtin_amdgcn_rcpf(1.0f + e);
            }
        }
    }
}

// ============================================================================
// Fallback fused kernel (no workspace needed): converts in-kernel via LDS.
// ============================================================================
#define XLD (D_ + 8)
__global__ __launch_bounds__(256, 1)
void rbf_fused_fallback(const float* __restrict__ x,
                        const float* __restrict__ centers,
                        const float* __restrict__ W1,
                        const float* __restrict__ b1,
                        const int*   __restrict__ sigma_p,
                        float* __restrict__ out) {
    __shared__ unsigned short xbf[BT][XLD];
    __shared__ unsigned short cbf[CB][XLD];
    __shared__ unsigned short phis[BT][PLD];
    __shared__ float x2s[BT];
    __shared__ float c2s[CB];

    const int tid   = threadIdx.x;
    const int wave  = tid >> 5;
    const int lane  = tid & 31;
    const int laneN = lane & 15;
    const int hi    = lane >> 4;
    const int m0    = blockIdx.x * BT;

    const float sig      = (float)sigma_p[0];
    const float negScale = -LOG2E * 0.5f / sig;

    if (tid < BT) x2s[tid] = 0.0f;
    __syncthreads();
    {
        const int r  = tid >> 3;
        const int c0 = (tid & 7) * 64;
        const v4f* xr = (const v4f*)(x + (size_t)(m0 + r) * D_ + c0);
        float s = 0.0f;
#pragma unroll 4
        for (int j = 0; j < 16; ++j) {
            v4f v = xr[j];
            s = fmaf(v[0], v[0], s); s = fmaf(v[1], v[1], s);
            s = fmaf(v[2], v[2], s); s = fmaf(v[3], v[3], s);
            unsigned* dst = (unsigned*)&xbf[r][c0 + j * 4];
            dst[0] = pack2bf(v[0], v[1]);
            dst[1] = pack2bf(v[2], v[3]);
        }
        atomicAdd(&x2s[r], s);
    }

    v8f acc2[2][8];
#pragma unroll
    for (int i = 0; i < 2; ++i)
#pragma unroll
        for (int j = 0; j < 8; ++j)
#pragma unroll
            for (int e = 0; e < 8; ++e) acc2[i][j][e] = 0.0f;

    const int mt = wave & 1;
    const int nt = wave >> 1;

    for (int cc = 0; cc < C_; cc += CB) {
        __syncthreads();
        if (tid < CB) c2s[tid] = 0.0f;
        __syncthreads();
        {
            const int r  = tid >> 2;
            const int k0 = (tid & 3) * 128;
            const v4f* cr = (const v4f*)(centers + (size_t)(cc + r) * D_ + k0);
            float s = 0.0f;
#pragma unroll 4
            for (int j = 0; j < 32; ++j) {
                v4f v = cr[j];
                s = fmaf(v[0], v[0], s); s = fmaf(v[1], v[1], s);
                s = fmaf(v[2], v[2], s); s = fmaf(v[3], v[3], s);
                unsigned* dst = (unsigned*)&cbf[r][k0 + j * 4];
                dst[0] = pack2bf(v[0], v[1]);
                dst[1] = pack2bf(v[2], v[3]);
            }
            atomicAdd(&c2s[r], s);
        }
        __syncthreads();

        v8f sacc;
#pragma unroll
        for (int e = 0; e < 8; ++e) sacc[e] = 0.0f;
        {
            const unsigned short* arow = &xbf[mt * 16 + laneN][hi * 16];
            const unsigned short* brow = &cbf[nt * 16 + laneN][hi * 16];
#pragma unroll 8
            for (int kk = 0; kk < D_; kk += 32) {
                v16us av = pack16(*(const v8us*)(arow + kk),
                                  *(const v8us*)(arow + kk + 8));
                v16us bv = pack16(*(const v8us*)(brow + kk),
                                  *(const v8us*)(brow + kk + 8));
                sacc = wmma_bf16(av, bv, sacc);
            }
        }
        {
            const float c2v = c2s[nt * 16 + laneN];
#pragma unroll
            for (int j = 0; j < 8; ++j) {
                const int m = mt * 16 + j + 8 * hi;
                float d2 = fmaxf(fmaf(-2.0f, sacc[j], x2s[m] + c2v), 0.0f);
                float d  = __builtin_amdgcn_sqrtf(d2);
                float p  = __builtin_amdgcn_exp2f(d * negScale);
                phis[m][nt * 16 + laneN] = f2bf(p);
            }
        }
        __syncthreads();

#pragma unroll
        for (int ks = 0; ks < 2; ++ks) {
            v16us afr[2];
#pragma unroll
            for (int mt2 = 0; mt2 < 2; ++mt2) {
                const unsigned short* ar = &phis[mt2 * 16 + laneN][ks * 32 + hi * 16];
                afr[mt2] = pack16(*(const v8us*)ar, *(const v8us*)(ar + 8));
            }
#pragma unroll
            for (int nt2 = 0; nt2 < 8; ++nt2) {
                const int hcol = wave * 128 + nt2 * 16 + laneN;
                const v4f* w4  = (const v4f*)(W1 + (size_t)hcol * C_ + cc
                                              + ks * 32 + hi * 16);
                v4f w0 = w4[0], w1v = w4[1], w2 = w4[2], w3 = w4[3];
                v8ui bu;
                bu[0] = pack2bf(w0[0], w0[1]); bu[1] = pack2bf(w0[2], w0[3]);
                bu[2] = pack2bf(w1v[0], w1v[1]); bu[3] = pack2bf(w1v[2], w1v[3]);
                bu[4] = pack2bf(w2[0], w2[1]); bu[5] = pack2bf(w2[2], w2[3]);
                bu[6] = pack2bf(w3[0], w3[1]); bu[7] = pack2bf(w3[2], w3[3]);
                v16us bfr = __builtin_bit_cast(v16us, bu);
                acc2[0][nt2] = wmma_bf16(afr[0], bfr, acc2[0][nt2]);
                acc2[1][nt2] = wmma_bf16(afr[1], bfr, acc2[1][nt2]);
            }
        }
    }

#pragma unroll
    for (int nt2 = 0; nt2 < 8; ++nt2) {
        const int hcol = wave * 128 + nt2 * 16 + laneN;
        const float bb = b1[hcol];
#pragma unroll
        for (int mt2 = 0; mt2 < 2; ++mt2) {
#pragma unroll
            for (int j = 0; j < 8; ++j) {
                const int row = m0 + mt2 * 16 + j + 8 * hi;
                float z = acc2[mt2][nt2][j] + bb;
                float e = __builtin_amdgcn_exp2f(-z * LOG2E);
                out[(size_t)row * H_ + hcol] = __builtin_amdgcn_rcpf(1.0f + e);
            }
        }
    }
}

// ============================================================================
extern "C" void kernel_launch(void* const* d_in, const int* in_sizes, int n_in,
                              void* d_out, int out_size, void* d_ws, size_t ws_size,
                              hipStream_t stream) {
    (void)in_sizes; (void)n_in; (void)out_size;
    const float* x       = (const float*)d_in[0];
    const float* centers = (const float*)d_in[1];
    const float* W1      = (const float*)d_in[2];
    const float* b1      = (const float*)d_in[3];
    const int*   sigma   = (const int*)d_in[4];
    float*       out     = (float*)d_out;

    if (d_ws != nullptr && ws_size >= WS_NEED) {
        char* ws = (char*)d_ws;
        unsigned short* xbf  = (unsigned short*)(ws + WS_XBF);
        unsigned short* cbf  = (unsigned short*)(ws + WS_CBF);
        unsigned short* w1bf = (unsigned short*)(ws + WS_W1BF);
        float* x2g = (float*)(ws + WS_X2);
        float* c2g = (float*)(ws + WS_C2);

        hipLaunchKernelGGL(conv_rows_kernel, dim3(B_ / 8), dim3(256), 0, stream,
                           x, xbf, x2g);
        hipLaunchKernelGGL(conv_rows_kernel, dim3(C_ / 8), dim3(256), 0, stream,
                           centers, cbf, c2g);
        hipLaunchKernelGGL(conv_flat_kernel,
                           dim3((size_t)H_ * C_ / 8 / 256), dim3(256), 0, stream,
                           W1, w1bf);
        hipLaunchKernelGGL(rbf_fused_fast, dim3(B_ / BT), dim3(256), 0, stream,
                           xbf, cbf, w1bf, x2g, c2g, b1, sigma, out);
    } else {
        hipLaunchKernelGGL(rbf_fused_fallback, dim3(B_ / BT), dim3(256), 0, stream,
                           x, centers, W1, b1, sigma, out);
    }
}
